// MultiHeadAttention_69999376990680
// MI455X (gfx1250) — matrix operand order repaired and checked by reading, not compiled, not run on hardware
//
#include <hip/hip_runtime.h>
#include <hip/hip_bf16.h>

typedef __attribute__((ext_vector_type(16))) _Float16 v16h;
typedef __attribute__((ext_vector_type(8)))  _Float16 v8h;
typedef __attribute__((ext_vector_type(4)))  _Float16 v4h;
typedef __attribute__((ext_vector_type(8)))  float    v8f;
typedef __attribute__((ext_vector_type(4)))  unsigned int u32x4;
typedef __attribute__((ext_vector_type(8)))  int i32x8;
typedef __attribute__((ext_vector_type(4)))  int i32x4;

#define WMMA_F16(A_, B_, C_) \
  __builtin_amdgcn_wmma_f32_16x16x32_f16(false, (A_), false, (B_), (short)0, (C_), false, false)

// ---------------------------------------------------------------------------
// Fragment loaders matching CDNA5 ISA §7.12.2 (wave32).
// A operand (16x32 f16): lane L holds row m=L%16; halves 0..7 = K=koff..koff+7,
// halves 8..15 = K=16+koff..16+koff+7 with koff = (L<16)?0:8. src row-major [M,K].
__device__ __forceinline__ v16h load_frag_a(const _Float16* __restrict__ src,
                                            int ld, int row, int k0) {
  const int lane = threadIdx.x & 31;
  const int koff = (lane & 16) ? 8 : 0;
  const _Float16* p = src + (size_t)(row + (lane & 15)) * ld + (k0 + koff);
  v8h lo = *(const v8h*)(p);
  v8h hi = *(const v8h*)(p + 16);
  v16h f;
#pragma unroll
  for (int i = 0; i < 8; ++i) { f[i] = lo[i]; f[8 + i] = hi[i]; }
  return f;
}

// B operand (32x16 f16): lane L holds column n=n0+L%16; elements 0..7 = K=koff..koff+7,
// 8..15 = K=16+koff..16+koff+7, koff=(L<16)?0:8.
// src must be K-major: base points at element [k0][n0], row stride ld (global memory).
__device__ __forceinline__ v16h load_frag_b(const _Float16* __restrict__ base, int ld) {
  const int lane = threadIdx.x & 31;
  const int koff = (lane & 16) ? 8 : 0;
  const _Float16* p = base + (size_t)koff * ld + (lane & 15);
  v16h f;
#pragma unroll
  for (int i = 0; i < 8; ++i) { f[i] = p[(size_t)i * ld]; f[8 + i] = p[(size_t)(16 + i) * ld]; }
  return f;
}

// B operand from LDS tile staged by the TDM. Row stride LDB halves (132 = 128 + 2-dword
// pad inserted by the TDM pad feature). Lane holds column L%16; elements are K rows
// koff..koff+7, 16+koff..16+koff+7 (koff=(L<16)?0:8), read as scalar halves.
__device__ __forceinline__ v16h load_frag_b_lds(const _Float16* p, int ldb) {
  const int lane = threadIdx.x & 31;
  const int koff = (lane & 16) ? 8 : 0;
  const _Float16* q = p + koff * ldb + (lane & 15);
  v16h f;
#pragma unroll
  for (int i = 0; i < 8; ++i) {
    f[i] = q[i * ldb]; f[8 + i] = q[(16 + i) * ldb];
  }
  return f;
}

// ---------------------------------------------------------------------------
// Tensor Data Mover: 2D tile load (global -> LDS) per CDNA5 ISA ch.8 D# layout.
// tile_w/tile_h/stride in 2-byte elements. Adds 2 dwords of LDS padding after each
// 128-half row (pad_interval code 5 = 64 dwords, pad_amount code 1 = 2 dwords).
__device__ __forceinline__ void tdm_load_tile_2d(unsigned lds_byte_addr,
                                                 const void* gsrc,
                                                 unsigned tile_w, unsigned tile_h,
                                                 unsigned stride_elems) {
  const unsigned long long ga = (unsigned long long)(size_t)gsrc;
  u32x4 g0;
  g0[0] = 1u;                                    // count=1 (valid user descriptor)
  g0[1] = lds_byte_addr;                         // lds_addr [63:32]
  g0[2] = (unsigned)(ga & 0xFFFFFFFFu);          // global_addr[31:0]
  g0[3] = (unsigned)((ga >> 32) & 0x1FFFFFFu)    // global_addr[56:32]
        | (2u << 30);                            // type = 2 ("image")
  i32x8 g1;
  g1[0] = (int)((1u << 16)                       // data_size = 1 -> 2 bytes
              | (1u << 20)                       // pad_enable
              | (5u << 22)                       // pad_interval: 64 dwords
              | (1u << 25));                     // pad_amount: 2 dwords
  g1[1] = (int)((tile_w & 0xFFFFu) << 16);                               // tensor_dim0 lo
  g1[2] = (int)(((tile_w >> 16) & 0xFFFFu) | ((tile_h & 0xFFFFu) << 16));// td0 hi | td1 lo
  g1[3] = (int)(((tile_h >> 16) & 0xFFFFu) | ((tile_w & 0xFFFFu) << 16));// td1 hi | tile_dim0
  g1[4] = (int)(tile_h & 0xFFFFu);                                       // tile_dim1 (tile_dim2=0)
  g1[5] = (int)stride_elems;                                             // tensor_dim0_stride lo
  g1[6] = 0;                                                             // stride0 hi | stride1 lo
  g1[7] = 0;
  const i32x4 gz4 = {0, 0, 0, 0};                // groups 2/3 unused (2D)
  const i32x8 gz8 = {0, 0, 0, 0, 0, 0, 0, 0};
  __builtin_amdgcn_tensor_load_to_lds(g0, g1, gz4, gz4, gz8, 0);
}

// ---------------------------------------------------------------------------
// Precision converts
__global__ void cvt_f32_to_f16(const float* __restrict__ in, _Float16* __restrict__ out,
                               long n) {
  long i = (long)blockIdx.x * blockDim.x + threadIdx.x;
  const long stride = (long)gridDim.x * blockDim.x;
  for (; i < n; i += stride) out[i] = (_Float16)in[i];
}

// in: [rows][cols] f32 row-major (torch weight [out,in]) -> out: [cols][rows] f16 (K-major)
__global__ void cvt_transpose_f32_to_f16(const float* __restrict__ in,
                                         _Float16* __restrict__ out, int rows, int cols) {
  const long total = (long)rows * cols;
  long i = (long)blockIdx.x * blockDim.x + threadIdx.x;
  const long stride = (long)gridDim.x * blockDim.x;
  for (; i < total; i += stride) {
    const int r = (int)(i / cols), c = (int)(i % cols);
    out[(size_t)c * rows + r] = (_Float16)in[i];
  }
}

// ---------------------------------------------------------------------------
// WMMA GEMM: C[M,Nout] = A[M,K] @ Bt[K,Nout] + bias.
// 128x128 block tile, 4 waves of 64x64, 16 v_wmma per K-step of 32.
// B tile (32x128) is staged global->LDS by the TDM, double-buffered so the DMA of
// tile k+32 overlaps the WMMAs on tile k. A fragments stream straight from global
// (fully-coalesced contiguous b128s, L2-resident reuse).
// EPI==0: scatter f16 into Qt [B,H,D,N], K [B,H,N,D], V [B,H,N,D]
// EPI==1: write f32 to outf
template<int EPI>
__global__ __launch_bounds__(128) void gemm_wmma(
    const _Float16* __restrict__ A,
    const _Float16* __restrict__ Bt,
    const float*    __restrict__ bias,
    _Float16* __restrict__ qt_ws, _Float16* __restrict__ k_ws, _Float16* __restrict__ v_ws,
    float* __restrict__ outf,
    int Mrows, int Nout, int K)
{
  (void)Mrows;
  constexpr int LDB = 132;            // halves per LDS row (128 + TDM pad)
  constexpr int BUFH = 32 * LDB;      // halves per buffer
  __shared__ __align__(16) _Float16 bsm[2 * BUFH];

  const int wave = threadIdx.x >> 5;
  const int lane = threadIdx.x & 31;
  const int rowBase = blockIdx.y * 128 + (wave >> 1) * 64;
  const int colTile = blockIdx.x * 128;            // block-wide B tile origin
  const int colBase = colTile + (wave & 1) * 64;
  const unsigned lds0 = (unsigned)(size_t)(void*)&bsm[0];

  // Preload first B tile.
  if (wave == 0) {
    tdm_load_tile_2d(lds0, Bt + colTile, 128, 32, (unsigned)Nout);
    __builtin_amdgcn_s_wait_tensorcnt(0);
  }
  __syncthreads();

  v8f acc[4][4] = {};
  int buf = 0;
  for (int k0 = 0; k0 < K; k0 += 32) {
    if (wave == 0 && (k0 + 32) < K)
      tdm_load_tile_2d(lds0 + (unsigned)((buf ^ 1) * BUFH * 2),
                       Bt + (size_t)(k0 + 32) * Nout + colTile, 128, 32, (unsigned)Nout);

    v16h af[4], bf[4];
#pragma unroll
    for (int i = 0; i < 4; ++i) af[i] = load_frag_a(A, K, rowBase + 16 * i, k0);
#pragma unroll
    for (int j = 0; j < 4; ++j)
      bf[j] = load_frag_b_lds(bsm + buf * BUFH + (wave & 1) * 64 + 16 * j, LDB);
#pragma unroll
    for (int i = 0; i < 4; ++i)
#pragma unroll
      for (int j = 0; j < 4; ++j)
        acc[i][j] = WMMA_F16(af[i], bf[j], acc[i][j]);

    if (wave == 0) __builtin_amdgcn_s_wait_tensorcnt(0);
    __syncthreads();
    buf ^= 1;
  }

  const int halfR = (lane & 16) ? 8 : 0;
  const int cn = lane & 15;
#pragma unroll
  for (int i = 0; i < 4; ++i) {
#pragma unroll
    for (int j = 0; j < 4; ++j) {
      const int cc = colBase + 16 * j + cn;
      const float bv = bias[cc];
#pragma unroll
      for (int g = 0; g < 8; ++g) {
        const int rr = rowBase + 16 * i + halfR + g;
        const float v = acc[i][j][g] + bv;
        if (EPI == 0) {
          const int which = cc >> 10;          // 0=Q 1=K 2=V
          const int rem = cc & 1023;
          const int h = rem >> 6, d = rem & 63;
          const int b = rr >> 11, nn = rr & 2047;
          const _Float16 hv = (_Float16)v;
          if (which == 0)      qt_ws[((size_t)(b * 16 + h) * 64 + d) * 2048 + nn] = hv;
          else if (which == 1) k_ws[((size_t)(b * 16 + h) * 2048 + nn) * 64 + d] = hv;
          else                 v_ws[((size_t)(b * 16 + h) * 2048 + nn) * 64 + d] = hv;
        } else {
          outf[(size_t)rr * Nout + cc] = v;
        }
      }
    }
  }
}

// ---------------------------------------------------------------------------
// Flash attention (wave32): each wave owns 16 queries, streams keys 32 at a time.
// Computes S^T = K·Q^T so the P accumulator lands directly in A-operand layout
// for the P·V WMMA (no LDS transpose, no cross-lane moves).
__global__ __launch_bounds__(128) void flash_attn(
    const _Float16* __restrict__ qt_ws,  // [B,H,D,N]
    const _Float16* __restrict__ k_ws,   // [B,H,N,D]
    const _Float16* __restrict__ v_ws,   // [B,H,N,D]
    _Float16* __restrict__ attn)         // [B,N,H*D]
{
  const int N = 2048, D = 64;
  const int bh = blockIdx.y;
  const int b = bh >> 4, h = bh & 15;
  const int wave = threadIdx.x >> 5;
  const int lane = threadIdx.x & 31;
  const int qbase = blockIdx.x * 64 + wave * 16;
  const int halfR = (lane & 16) ? 8 : 0;

  const _Float16* Qt = qt_ws + (size_t)bh * D * N;
  const _Float16* Kp = k_ws + (size_t)bh * N * D;
  const _Float16* Vp = v_ws + (size_t)bh * N * D;

  // Q^T fragments for this wave's 16 queries (B operand, d rows = lane)
  const v16h qb0 = load_frag_b(Qt + qbase, N);                     // d = 0..31
  const v16h qb1 = load_frag_b(Qt + (size_t)32 * N + qbase, N);    // d = 32..63

  v8f o[4] = {};
  float rowM = -3.0e38f;
  float rowL = 0.0f;
  const float c1 = 0.125f * 1.44269504088896340736f;  // (1/sqrt(D)) * log2(e)

  for (int j = 0; j < N; j += 32) {
    // S^T tiles: rows = keys, cols = queries
    v8f s0 = {}, s1 = {};
    s0 = WMMA_F16(load_frag_a(Kp, D, j, 0),       qb0, s0);
    s0 = WMMA_F16(load_frag_a(Kp, D, j, 32),      qb1, s0);
    s1 = WMMA_F16(load_frag_a(Kp, D, j + 16, 0),  qb0, s1);
    s1 = WMMA_F16(load_frag_a(Kp, D, j + 16, 32), qb1, s1);

    // per-lane: 16 scores for query m=lane%16, keys {koff+g, 16+koff+g}
    float sv[16];
#pragma unroll
    for (int g = 0; g < 8; ++g) { sv[g] = s0[g] * c1; sv[8 + g] = s1[g] * c1; }
    float tmax = sv[0];
#pragma unroll
    for (int i = 1; i < 16; ++i) tmax = fmaxf(tmax, sv[i]);
    tmax = fmaxf(tmax, __shfl_xor(tmax, 16, 32));   // other half holds other 16 keys
    const float newM = fmaxf(rowM, tmax);

    float p[16];
    float ps = 0.0f;
#pragma unroll
    for (int i = 0; i < 16; ++i) { p[i] = exp2f(sv[i] - newM); ps += p[i]; }
    ps += __shfl_xor(ps, 16, 32);
    const float resc = exp2f(rowM - newM);
    rowL = rowL * resc + ps;
    rowM = newM;

    // rescale O accumulators (rows m = halfR+g; stats live in lane halfR+g)
#pragma unroll
    for (int g = 0; g < 8; ++g) {
      const float fg = __shfl(resc, halfR + g, 32);
#pragma unroll
      for (int t = 0; t < 4; ++t) o[t][g] *= fg;
    }

    // P is already in A-operand layout: halves 0..7 <- s0 accs, 8..15 <- s1 accs
    v16h pa;
#pragma unroll
    for (int i = 0; i < 16; ++i) pa[i] = (_Float16)p[i];

    // O += P @ V  (V rows = keys = lane, columns = 16 d's per tile)
#pragma unroll
    for (int t = 0; t < 4; ++t) {
      const v16h vb = load_frag_b(Vp + (size_t)j * D + 16 * t, D);
      o[t] = WMMA_F16(pa, vb, o[t]);
    }
  }

  const float linv = 1.0f / rowL;
  _Float16* outp = attn + ((size_t)b * N + qbase) * 1024 + h * 64;
#pragma unroll
  for (int g = 0; g < 8; ++g) {
    const float lg = __shfl(linv, halfR + g, 32);
    const int m = halfR + g;
#pragma unroll
    for (int t = 0; t < 4; ++t)
      outp[(size_t)m * 1024 + 16 * t + (lane & 15)] = (_Float16)(o[t][g] * lg);
  }
}

// ---------------------------------------------------------------------------
extern "C" void kernel_launch(void* const* d_in, const int* in_sizes, int n_in,
                              void* d_out, int out_size, void* d_ws, size_t ws_size,
                              hipStream_t stream) {
  (void)in_sizes; (void)n_in; (void)out_size; (void)ws_size;
  const float* x     = (const float*)d_in[0];  // [4,2048,1024]
  const float* Wqkv  = (const float*)d_in[1];  // [3072,1024]
  const float* bqkv  = (const float*)d_in[2];  // [3072]
  const float* Wproj = (const float*)d_in[3];  // [1024,1024]
  const float* bproj = (const float*)d_in[4];  // [1024]

  const size_t XN    = (size_t)4 * 2048 * 1024;  // 8388608
  const size_t WQKVN = (size_t)3072 * 1024;
  const size_t WPRJN = (size_t)1024 * 1024;

  _Float16* ws = (_Float16*)d_ws;
  _Float16* xh     = ws; ws += XN;
  _Float16* wqkvT  = ws; ws += WQKVN;   // [K=1024, 3072]
  _Float16* wprojT = ws; ws += WPRJN;   // [K=1024, 1024]
  _Float16* qt_ws  = ws; ws += XN;      // [B,H,D,N]
  _Float16* k_ws   = ws; ws += XN;      // [B,H,N,D]
  _Float16* v_ws   = ws; ws += XN;      // [B,H,N,D]
  _Float16* attn   = ws; ws += XN;      // [B,N,H*D]

  cvt_f32_to_f16<<<1024, 256, 0, stream>>>(x, xh, (long)XN);
  cvt_transpose_f32_to_f16<<<1024, 256, 0, stream>>>(Wqkv, wqkvT, 3072, 1024);
  cvt_transpose_f32_to_f16<<<512, 256, 0, stream>>>(Wproj, wprojT, 1024, 1024);

  dim3 blk(128);
  // QKV: M=8192, Nout=3072, K=1024
  gemm_wmma<0><<<dim3(24, 64), blk, 0, stream>>>(xh, wqkvT, bqkv,
                                                 qt_ws, k_ws, v_ws, nullptr,
                                                 8192, 3072, 1024);
  // Attention: (N/64, B*H)
  flash_attn<<<dim3(32, 64), blk, 0, stream>>>(qt_ws, k_ws, v_ws, attn);
  // Proj: M=8192, Nout=1024, K=1024 -> f32 out
  gemm_wmma<1><<<dim3(8, 64), blk, 0, stream>>>(attn, wprojT, bproj,
                                                nullptr, nullptr, nullptr,
                                                (float*)d_out, 8192, 1024, 1024);
}
